// FusedQKVAttention_43817256354369
// MI455X (gfx1250) — compile-verified, matrix operand order checked by reading, not run on hardware
//
#include <hip/hip_runtime.h>
#include <hip/hip_bf16.h>

#define EMBED   2048
#define NHEADS  16
#define HDIM    128
#define BATCH   2
#define SEQ     2048
#define BH      (BATCH * NHEADS)
#define MROWS   (BATCH * SEQ)      // 4096
#define NQKV    (3 * EMBED)        // 6144
#define TLD     40                 // padded LDS row stride in f16 (80 B, 16B-aligned)
#define NSTAGE  (EMBED / 32)       // 64 k-steps

typedef __attribute__((ext_vector_type(16))) _Float16 v16h;
typedef __attribute__((ext_vector_type(8)))  _Float16 v8h;
typedef __attribute__((ext_vector_type(4)))  _Float16 v4h;
typedef __attribute__((ext_vector_type(8)))  float    v8f;
typedef __attribute__((ext_vector_type(4)))  float    v4f;
typedef __attribute__((ext_vector_type(4)))  unsigned int u32x4;
typedef __attribute__((ext_vector_type(8)))  int      i32x8;
typedef __attribute__((ext_vector_type(4)))  int      i32x4;

#if __has_builtin(__builtin_amdgcn_tensor_load_to_lds)
#define HAVE_TDM 1
#else
#define HAVE_TDM 0
#endif

__device__ __forceinline__ v8f wmma16(v16h a, v16h b, v8f c) {
  // D = A(16x32 f16) * B(32x16 f16) + C(16x16 f32)
  return __builtin_amdgcn_wmma_f32_16x16x32_f16(
      false, a, false, b, (short)0, c, false, false);
}

// 16x32 f16 WMMA A-fragment (identical per-lane pattern serves as B-fragment,
// with "row" meaning the B column).  ISA 7.12.2: lane L -> row L&15,
// K elems (L<16 ? 0..7 : 8..15) and +16.  Two 16-byte loads per lane.
__device__ __forceinline__ v16h load_frag(const _Float16* base, int ld) {
  const int lane = threadIdx.x & 31;
  const int r  = lane & 15;
  const int kb = (lane >> 4) << 3;
  const _Float16* p = base + (size_t)r * ld + kb;
  v8h lo = *(const v8h*)(p);
  v8h hi = *(const v8h*)(p + 16);
  v16h f;
#pragma unroll
  for (int i = 0; i < 8; ++i) { f[i] = lo[i]; f[i + 8] = hi[i]; }
  return f;
}

#if HAVE_TDM
// Issue one TDM load of a 128x32 f16 tile (row stride EMBED elements) into
// LDS at byte offset ldsOff.  Hardware pads +16B after every 64B row so the
// LDS image has the TLD=40-halfword padded stride.  D# per ISA ch.8:
//   group0: count=1 | lds_addr | global_addr(57b) | type=2
//   group1: data_size=2B, pad_enable, pad_interval=16dw, pad_amount=4dw,
//           tensor_dim0=EMBED, tensor_dim1=rows, tile=32x128, stride0=EMBED
__device__ __forceinline__ void tdm_load_tile(const _Float16* tile,
                                              unsigned ldsOff, int tensorRows) {
  unsigned long long ga = (unsigned long long)(uintptr_t)tile;
  u32x4 g0;
  g0[0] = 1u;                                    // count=1, user mode
  g0[1] = ldsOff;                                // LDS byte address
  g0[2] = (unsigned)ga;                          // global_addr[31:0]
  g0[3] = (unsigned)(ga >> 32) | 0x80000000u;    // addr[56:32] | type=2<<30
  i32x8 g1;
  g1[0] = (1 << 16) | (1 << 20) | (3 << 22) | (3 << 25);
  g1[1] = (int)((unsigned)(EMBED & 0xffff) << 16);            // dim0[15:0]
  g1[2] = (int)((EMBED >> 16) | ((unsigned)(tensorRows & 0xffff) << 16));
  g1[3] = (int)(((unsigned)tensorRows >> 16) | (32u << 16));  // tile_dim0=32
  g1[4] = 128;                                                // tile_dim1=128
  g1[5] = EMBED;                                              // dim0_stride lo
  g1[6] = 0;
  g1[7] = 0;
  i32x4 z4 = (i32x4)0;
#if defined(__clang_major__) && (__clang_major__ >= 23)
  i32x8 z8 = (i32x8)0;
  __builtin_amdgcn_tensor_load_to_lds(g0, g1, z4, z4, z8, 0);
#else
  __builtin_amdgcn_tensor_load_to_lds(g0, g1, z4, z4, 0);
#endif
}
#endif  // HAVE_TDM

// ---------------------------------------------------------------------------
// Kernel 0: f32 -> f16 convert (bandwidth-bound, ~4us total at 23.3 TB/s).
// Makes the GEMM tiles pure f16 so the TDM can stage them byte-for-byte.
// ---------------------------------------------------------------------------
__global__ __launch_bounds__(256) void cvt_f16_kernel(
    const float* __restrict__ in, _Float16* __restrict__ out, int n4) {
  int i = blockIdx.x * 256 + threadIdx.x;
  if (i < n4) {
    v4f v = *(const v4f*)(in + (size_t)4 * i);
    v4h h;
#pragma unroll
    for (int q = 0; q < 4; ++q) h[q] = (_Float16)v[q];
    *(v4h*)(out + (size_t)4 * i) = h;
  }
}

// ---------------------------------------------------------------------------
// TDM-staged f16 GEMM:  D[m,n] = sum_k A[m,k] * B[n,k] + bias[n]
// 128x128 block tile, K-step 32, double-buffered LDS fed by TENSOR_LOAD_TO_LDS
// (wave 0 programs the D#; TENSORcnt/s_wait_tensorcnt + barrier publishes).
// EPI==0: scatter into Q [BH,S,128], K [BH,S,128], V transposed [BH,128,S].
// EPI==1: f32 output [M, EMBED] + bias.
// ---------------------------------------------------------------------------
template <int EPI>
__global__ __launch_bounds__(256) void gemm_tdm_kernel(
    const _Float16* __restrict__ A,   // [Mtot, 2048] f16
    const _Float16* __restrict__ Bm,  // [Ntot, 2048] f16
    const float* __restrict__ bias,   // [Ntot]
    int Mtot, int Ntot,
    _Float16* __restrict__ Qb, _Float16* __restrict__ Kb,
    _Float16* __restrict__ Vt, float* __restrict__ Out) {
  __shared__ _Float16 Asm[2][128 * TLD];
  __shared__ _Float16 Bsm[2][128 * TLD];

  const int tid = threadIdx.x;
  const int w   = tid >> 5;
  const int wm  = w & 3;               // 4 M slots of 32 rows
  const int wn  = w >> 2;              // 2 N slots of 64 cols
  const int m0  = blockIdx.x * 128;
  const int n0  = blockIdx.y * 128;

  v8f acc[2][4];
#pragma unroll
  for (int i = 0; i < 2; ++i)
#pragma unroll
    for (int j = 0; j < 4; ++j) acc[i][j] = (v8f)0.0f;

#if HAVE_TDM
  const unsigned aOff0 = (unsigned)(uintptr_t)&Asm[0][0];
  const unsigned aOff1 = (unsigned)(uintptr_t)&Asm[1][0];
  const unsigned bOff0 = (unsigned)(uintptr_t)&Bsm[0][0];
  const unsigned bOff1 = (unsigned)(uintptr_t)&Bsm[1][0];
  if (w == 0) {  // prologue: stage 0 into buffer 0
    tdm_load_tile(A  + (size_t)m0 * EMBED, aOff0, Mtot);
    tdm_load_tile(Bm + (size_t)n0 * EMBED, bOff0, Ntot);
  }
  for (int s = 0; s < NSTAGE; ++s) {
    __syncthreads();                   // readers of buf (s+1)&1 are done
    if (w == 0) {
      if (s + 1 < NSTAGE) {
        const unsigned ao = ((s + 1) & 1) ? aOff1 : aOff0;
        const unsigned bo = ((s + 1) & 1) ? bOff1 : bOff0;
        tdm_load_tile(A  + (size_t)m0 * EMBED + (s + 1) * 32, ao, Mtot);
        tdm_load_tile(Bm + (size_t)n0 * EMBED + (s + 1) * 32, bo, Ntot);
        __builtin_amdgcn_s_wait_tensorcnt(2);   // oldest stage (s) landed
      } else {
        __builtin_amdgcn_s_wait_tensorcnt(0);
      }
    }
    __syncthreads();                   // publish stage s to all waves
    const _Float16* Ab = &Asm[s & 1][0];
    const _Float16* Bb = &Bsm[s & 1][0];
#else
  for (int s = 0; s < NSTAGE; ++s) {
    __syncthreads();
#pragma unroll
    for (int it = 0; it < 2; ++it) {   // cooperative fallback copy (f16)
      int e = (it * 256 + tid) * 8;
      int r = e >> 5, c = e & 31;
      *(v8h*)(&Asm[0][0] + r * TLD + c) =
          *(const v8h*)(A + (size_t)(m0 + r) * EMBED + s * 32 + c);
      *(v8h*)(&Bsm[0][0] + r * TLD + c) =
          *(const v8h*)(Bm + (size_t)(n0 + r) * EMBED + s * 32 + c);
    }
    __syncthreads();
    const _Float16* Ab = &Asm[0][0];
    const _Float16* Bb = &Bsm[0][0];
#endif
    v16h af[2], bf[4];
#pragma unroll
    for (int i = 0; i < 2; ++i) af[i] = load_frag(Ab + (wm * 32 + i * 16) * TLD, TLD);
#pragma unroll
    for (int j = 0; j < 4; ++j) bf[j] = load_frag(Bb + (wn * 64 + j * 16) * TLD, TLD);
#pragma unroll
    for (int i = 0; i < 2; ++i)
#pragma unroll
      for (int j = 0; j < 4; ++j) acc[i][j] = wmma16(af[i], bf[j], acc[i][j]);
  }

  // Epilogue: D layout — col n = lane&15, rows m = (lane>=16?8:0)+t
  const int lane = tid & 31;
  const int nloc = lane & 15;
  const int mg   = lane >> 4;
  if (EPI == 0) {
    const int which = n0 >> 11;                 // 0=Q, 1=K, 2=V
    const int h     = (n0 >> 7) & (NHEADS - 1);
#pragma unroll
    for (int i = 0; i < 2; ++i) {
      const int mbase = m0 + wm * 32 + i * 16 + mg * 8;
      const int b  = mbase >> 11;               // / SEQ
      const int s0 = mbase & (SEQ - 1);
      const int bh = b * NHEADS + h;
#pragma unroll
      for (int j = 0; j < 4; ++j) {
        const int d = wn * 64 + j * 16 + nloc;  // 0..127 within the head
        const float bv = bias[n0 + d];
        if (which == 2) {
          v8h ph;                               // consecutive s -> 16B store
#pragma unroll
          for (int t = 0; t < 8; ++t) ph[t] = (_Float16)(acc[i][j][t] + bv);
          *(v8h*)(Vt + ((size_t)bh * HDIM + d) * SEQ + s0) = ph;
        } else {
          _Float16* dst = (which == 0 ? Qb : Kb) + ((size_t)bh * SEQ + s0) * HDIM + d;
#pragma unroll
          for (int t = 0; t < 8; ++t) dst[(size_t)t * HDIM] = (_Float16)(acc[i][j][t] + bv);
        }
      }
    }
  } else {
#pragma unroll
    for (int i = 0; i < 2; ++i) {
      const int mbase = m0 + wm * 32 + i * 16 + mg * 8;
#pragma unroll
      for (int j = 0; j < 4; ++j) {
        const int n = n0 + wn * 64 + j * 16 + nloc;
        const float bv = bias[n];
#pragma unroll
        for (int t = 0; t < 8; ++t)
          Out[(size_t)(mbase + t) * EMBED + n] = acc[i][j][t] + bv;
      }
    }
  }
}

// ---------------------------------------------------------------------------
// Causal flash attention. One wave owns 16 q-rows; Q and the 16x128 f32 O
// accumulator stay in registers; K and transposed-V B-fragments stream from
// global (per-(b,h) K/V = 1 MB -> resident in the 192 MB L2).
// ---------------------------------------------------------------------------
__global__ __launch_bounds__(256) void attn_kernel(
    const _Float16* __restrict__ Qb,   // [BH, SEQ, HDIM]
    const _Float16* __restrict__ Kb,   // [BH, SEQ, HDIM]
    const _Float16* __restrict__ Vt,   // [BH, HDIM, SEQ]
    _Float16* __restrict__ Oh) {       // [B, SEQ, EMBED]
  __shared__ _Float16 Pl[8][16 * TLD];        // per-wave private P bounce tile

  const int tid  = threadIdx.x;
  const int w    = tid >> 5;
  const int lane = tid & 31;
  const int nloc = lane & 15;
  const int mg   = lane >> 4;
  const int bh   = blockIdx.y;
  const int q0   = blockIdx.x * 128 + w * 16;
  const float scale = 0.08838834764831845f;   // 1/sqrt(128)

  const _Float16* Qp    = Qb + ((size_t)bh * SEQ + q0) * HDIM;
  const _Float16* Kbase = Kb + (size_t)bh * SEQ * HDIM;
  const _Float16* Vbase = Vt + (size_t)bh * HDIM * SEQ;
  _Float16* Pw = &Pl[w][0];

  v16h qf[4];
#pragma unroll
  for (int c = 0; c < 4; ++c) qf[c] = load_frag(Qp + c * 32, HDIM);

  v8f o[8];
#pragma unroll
  for (int f = 0; f < 8; ++f) o[f] = (v8f)0.0f;
  float mrow[8], lrow[8];
#pragma unroll
  for (int t = 0; t < 8; ++t) { mrow[t] = -3.0e38f; lrow[t] = 0.0f; }

  for (int kb = 0; kb < q0 + 16; kb += 32) {
    v8f s0 = (v8f)0.0f, s1 = (v8f)0.0f;
#pragma unroll
    for (int c = 0; c < 4; ++c) {
      v16h kf0 = load_frag(Kbase + (size_t)(kb)      * HDIM + c * 32, HDIM);
      v16h kf1 = load_frag(Kbase + (size_t)(kb + 16) * HDIM + c * 32, HDIM);
      s0 = wmma16(qf[c], kf0, s0);
      s1 = wmma16(qf[c], kf1, s1);
    }

    float mnew[8];
#pragma unroll
    for (int t = 0; t < 8; ++t) {
      const int q   = q0 + mg * 8 + t;
      const int kp0 = kb + nloc;
      const int kp1 = kb + 16 + nloc;
      float a = (kp0 <= q) ? s0[t] * scale : -3.0e38f;
      float b = (kp1 <= q) ? s1[t] * scale : -3.0e38f;
      s0[t] = a; s1[t] = b;
      mnew[t] = fmaxf(a, b);
    }
#pragma unroll
    for (int off = 1; off < 16; off <<= 1)
#pragma unroll
      for (int t = 0; t < 8; ++t)
        mnew[t] = fmaxf(mnew[t], __shfl_xor(mnew[t], off, 32));

    float psum[8];
#pragma unroll
    for (int t = 0; t < 8; ++t) {
      const float mn    = fmaxf(mrow[t], mnew[t]);
      const float alpha = __expf(mrow[t] - mn);
      mrow[t] = mn;
      lrow[t] *= alpha;
#pragma unroll
      for (int f = 0; f < 8; ++f) o[f][t] *= alpha;
      const float p0 = __expf(s0[t] - mn);
      const float p1 = __expf(s1[t] - mn);
      s0[t] = p0; s1[t] = p1;
      psum[t] = p0 + p1;
    }
#pragma unroll
    for (int off = 1; off < 16; off <<= 1)
#pragma unroll
      for (int t = 0; t < 8; ++t)
        psum[t] += __shfl_xor(psum[t], off, 32);
#pragma unroll
    for (int t = 0; t < 8; ++t) lrow[t] += psum[t];

    // P: D layout -> A layout via per-wave private LDS bounce
#pragma unroll
    for (int t = 0; t < 8; ++t) {
      const int m = mg * 8 + t;
      Pw[m * TLD + nloc]      = (_Float16)s0[t];
      Pw[m * TLD + 16 + nloc] = (_Float16)s1[t];
    }
    asm volatile("s_wait_dscnt 0" ::: "memory");   // same-wave DS RAW
    v16h pf = load_frag(Pw, TLD);

#pragma unroll
    for (int f = 0; f < 8; ++f) {
      v16h vf = load_frag(Vbase + (size_t)(f * 16) * SEQ + kb, SEQ);
      o[f] = wmma16(pf, vf, o[f]);
    }
  }

  const int b = bh / NHEADS;
  const int h = bh % NHEADS;
  float inv[8];
#pragma unroll
  for (int t = 0; t < 8; ++t) inv[t] = 1.0f / lrow[t];
#pragma unroll
  for (int f = 0; f < 8; ++f) {
#pragma unroll
    for (int t = 0; t < 8; ++t) {
      const int s = q0 + mg * 8 + t;
      const int d = f * 16 + nloc;
      Oh[((size_t)(b * SEQ + s)) * EMBED + h * HDIM + d] = (_Float16)(o[f][t] * inv[t]);
    }
  }
}

// ---------------------------------------------------------------------------
extern "C" void kernel_launch(void* const* d_in, const int* in_sizes, int n_in,
                              void* d_out, int out_size, void* d_ws, size_t ws_size,
                              hipStream_t stream) {
  const float* x    = (const float*)d_in[0];   // [2, 2048, 2048]
  const float* Wqkv = (const float*)d_in[1];   // [6144, 2048]
  const float* bqkv = (const float*)d_in[2];   // [6144]
  const float* Wout = (const float*)d_in[3];   // [2048, 2048]
  const float* bout = (const float*)d_in[4];   // [2048]
  // d_in[5] = attn_mask (bool causal) — causality is computed analytically.
  (void)in_sizes; (void)n_in; (void)out_size; (void)ws_size;

  const size_t nX  = (size_t)MROWS * EMBED;    // 8.39M
  const size_t nWq = (size_t)NQKV * EMBED;     // 12.58M
  const size_t nWo = (size_t)EMBED * EMBED;    // 4.19M
  const size_t nT  = (size_t)BH * SEQ * HDIM;  // 8.39M

  _Float16* Xh  = (_Float16*)d_ws;             // ~117 MB total f16 workspace
  _Float16* Wh  = Xh + nX;
  _Float16* Woh = Wh + nWq;
  _Float16* Qb  = Woh + nWo;
  _Float16* Kb  = Qb + nT;
  _Float16* Vt  = Kb + nT;
  _Float16* Oh  = Vt + nT;
  float* Out = (float*)d_out;

  dim3 blk(256);
  cvt_f16_kernel<<<(int)(nX  / 1024), blk, 0, stream>>>(x,    Xh,  (int)(nX  / 4));
  cvt_f16_kernel<<<(int)(nWq / 1024), blk, 0, stream>>>(Wqkv, Wh,  (int)(nWq / 4));
  cvt_f16_kernel<<<(int)(nWo / 1024), blk, 0, stream>>>(Wout, Woh, (int)(nWo / 4));

  gemm_tdm_kernel<0><<<dim3(MROWS / 128, NQKV / 128), blk, 0, stream>>>(
      Xh, Wh, bqkv, MROWS, NQKV, Qb, Kb, Vt, nullptr);
  attn_kernel<<<dim3(SEQ / 128, BH), blk, 0, stream>>>(Qb, Kb, Vt, Oh);
  gemm_tdm_kernel<1><<<dim3(MROWS / 128, EMBED / 128), blk, 0, stream>>>(
      Oh, Woh, bout, MROWS, EMBED, nullptr, nullptr, nullptr, Out);
}
